// DiscriminativeLoss_46420006535340
// MI455X (gfx1250) — matrix operand order repaired and checked by reading, not compiled, not run on hardware
//
#include <hip/hip_runtime.h>
#include <math.h>

// ---------------------------------------------------------------------------
// DiscriminativeLoss for MI455X (gfx1250, wave32)
//   embedded: (B=4, C=64, N=262144) f32, labels: (B,N) i32, K=32 clusters
//   Bandwidth-bound: 2 streaming passes over 256MB -> ~22us roofline @23.3TB/s
//   WMMA used for the 32x32 centroid Gram matrix (L_dist term).
// ---------------------------------------------------------------------------

#define BATCH 4
#define CHAN  64
#define NPTS  262144
#define KCL   32
#define NBLK  128                 // chunks per batch
#define CHUNK (NPTS / NBLK)       // 2048 points per block
#define TPB   256
#define PPT   (CHUNK / TPB)       // 8 points per thread

#define DELTA_D 1.5f
#define DELTA_V 0.5f

typedef __attribute__((ext_vector_type(16))) _Float16 v16h;
typedef __attribute__((ext_vector_type(8)))  float    v8f;
typedef __attribute__((ext_vector_type(4)))  float    v4f;
typedef __attribute__((ext_vector_type(4)))  int      v4i;

// workspace layout (floats)
#define OFF_CNT 0
#define OFF_SUM (OFF_CNT + BATCH*KCL)           // per-cluster channel sums [b][c][k]
#define OFF_MU  (OFF_SUM + BATCH*CHAN*KCL)      // centroids, [b][c][k]
#define OFF_V   (OFF_MU  + BATCH*CHAN*KCL)      // L_var segment sums [b][k]
#define OFF_LD  (OFF_V   + BATCH*KCL)           // per-batch L_dist
#define OFF_LR  (OFF_LD  + BATCH)               // per-batch L_reg
#define WS_FLOATS (OFF_LR + BATCH)

__global__ void init_ws_kernel(float* __restrict__ ws) {
    int i = blockIdx.x * blockDim.x + threadIdx.x;
    if (i < WS_FLOATS) ws[i] = 0.0f;
}

// ---------------------------------------------------------------------------
// Pass 1: per-cluster counts and channel sums.
// LDS accumulator layout [c][k] -> bank = (c*32+k)%64: 32 lanes with distinct
// cluster ids hit distinct banks. Forward chunk order (fills L2 for pass 2).
// ---------------------------------------------------------------------------
__global__ void __launch_bounds__(TPB)
seg_sum_kernel(const float* __restrict__ emb, const int* __restrict__ labels,
               float* __restrict__ ws) {
    __shared__ float lsum[CHAN * KCL];
    __shared__ float lcnt[KCL];

    const int b     = blockIdx.x / NBLK;
    const int chunk = blockIdx.x % NBLK;
    const int tid   = threadIdx.x;

    for (int i = tid; i < CHAN * KCL; i += TPB) lsum[i] = 0.0f;
    if (tid < KCL) lcnt[tid] = 0.0f;
    __syncthreads();

    const int n0 = chunk * CHUNK + tid * PPT;
    const int* lp = labels + (size_t)b * NPTS + n0;
    v4i li0 = *(const v4i*)(lp);
    v4i li1 = *(const v4i*)(lp + 4);
    int lab[PPT] = {li0.x, li0.y, li0.z, li0.w, li1.x, li1.y, li1.z, li1.w};

#pragma unroll
    for (int p = 0; p < PPT; ++p) atomicAdd(&lcnt[lab[p]], 1.0f);

    const float* base = emb + (size_t)b * CHAN * NPTS + n0;
    for (int c = 0; c < CHAN; ++c) {
        const float* xp = base + (size_t)c * NPTS;
        if (c + 1 < CHAN) __builtin_prefetch(xp + NPTS, 0, 1);  // next channel row
        v4f x0 = *(const v4f*)(xp);
        v4f x1 = *(const v4f*)(xp + 4);
        float xv[PPT] = {x0.x, x0.y, x0.z, x0.w, x1.x, x1.y, x1.z, x1.w};
        float* row = &lsum[c * KCL];
#pragma unroll
        for (int p = 0; p < PPT; ++p) atomicAdd(&row[lab[p]], xv[p]);
    }
    __syncthreads();

    float* gsum = ws + OFF_SUM + (size_t)b * CHAN * KCL;
    for (int i = tid; i < CHAN * KCL; i += TPB) atomicAdd(&gsum[i], lsum[i]);
    if (tid < KCL) atomicAdd(&ws[OFF_CNT + b * KCL + tid], lcnt[tid]);
}

// ---------------------------------------------------------------------------
// Centroids + L_dist (WMMA Gram) + L_reg. One wave32 per batch; EXEC all-1s
// through the WMMA region. G = Mu(32x64) x Mu^T(64x32) as 2x2 tiles of
// v_wmma_f32_16x16x32_f16, K=64 split in two K=32 steps. Exact f32 diagonal
// norms recover pairwise distances: ||mu_i - mu_j||^2 = n_i + n_j - 2 G_ij.
// ---------------------------------------------------------------------------
__global__ void __launch_bounds__(32)
centroid_dist_kernel(float* __restrict__ ws) {
    __shared__ _Float16 muh[KCL * CHAN];   // [k][c] f16 copy for WMMA
    __shared__ float    nrm[KCL];
    __shared__ float    red[KCL];

    const int b    = blockIdx.x;
    const int lane = threadIdx.x;          // 0..31 == cluster id

    const float cnt = ws[OFF_CNT + b * KCL + lane];
    const float inv = (cnt > 0.0f) ? (1.0f / cnt) : 1.0f;   // safe_counts
    const float* gsum = ws + OFF_SUM + (size_t)b * CHAN * KCL;
    float*       gmu  = ws + OFF_MU  + (size_t)b * CHAN * KCL;

    float nk = 0.0f;
    for (int c = 0; c < CHAN; ++c) {
        float m = gsum[c * KCL + lane] * inv;
        muh[lane * CHAN + c] = (_Float16)m;
        gmu[c * KCL + lane]  = m;          // [c][k] layout for pass 2
        nk += m * m;
    }
    nrm[lane] = nk;
    __syncthreads();

    const int mrow = lane & 15;
    const int hi   = (lane >= 16) ? 1 : 0;

    v8f c00 = {}, c01 = {}, c10 = {}, c11 = {};
#pragma unroll
    for (int s = 0; s < 2; ++s) {          // K blocks: channels [32s, 32s+32)
        v16h a0, a1, b0, b1;
#pragma unroll
        for (int h = 0; h < 16; ++h) {
            // A (16-bit 16x32): lanes0-15 K={0..7,16..23}; lanes16-31 K={8..15,24..31}
            int ka = h + ((h >= 8) ? 8 : 0) + (hi ? 8 : 0);
            // B (16-bit 32x16): lanes0-15 K={0..7,16..23}; lanes16-31 K={8..15,24..31}
            int kb = h + ((h >= 8) ? 8 : 0) + (hi ? 8 : 0);
            a0[h] = muh[(0 * 16 + mrow) * CHAN + s * 32 + ka];
            a1[h] = muh[(1 * 16 + mrow) * CHAN + s * 32 + ka];
            b0[h] = muh[(0 * 16 + mrow) * CHAN + s * 32 + kb];
            b1[h] = muh[(1 * 16 + mrow) * CHAN + s * 32 + kb];
        }
        c00 = __builtin_amdgcn_wmma_f32_16x16x32_f16(false, a0, false, b0, (short)0, c00, false, false);
        c01 = __builtin_amdgcn_wmma_f32_16x16x32_f16(false, a0, false, b1, (short)0, c01, false, false);
        c10 = __builtin_amdgcn_wmma_f32_16x16x32_f16(false, a1, false, b0, (short)0, c10, false, false);
        c11 = __builtin_amdgcn_wmma_f32_16x16x32_f16(false, a1, false, b1, (short)0, c11, false, false);
    }

    // C/D layout: lane<16 -> M=r, lane>=16 -> M=8+r; N = lane&15.
    float acc = 0.0f;
    const int nn = lane & 15;
#pragma unroll
    for (int r = 0; r < 8; ++r) {
        const int m = r + hi * 8;
#define HINGE(Cv, IT, JT)                                                    \
        {                                                                    \
            int i = (IT) * 16 + m, j = (JT) * 16 + nn;                       \
            float pn2 = nrm[i] + nrm[j] - 2.0f * (Cv)[r];                    \
            float pn  = (pn2 > 0.0f) ? sqrtf(pn2) : 0.0f;                    \
            if (i != j) {                                                    \
                float t = 2.0f * DELTA_D - pn;                               \
                if (t > 0.0f) acc += t * t;                                  \
            }                                                                \
        }
        HINGE(c00, 0, 0) HINGE(c01, 0, 1) HINGE(c10, 1, 0) HINGE(c11, 1, 1)
#undef HINGE
    }

    red[lane] = acc;
    __syncthreads();
    if (lane == 0) {
        float s = 0.0f;
        for (int i = 0; i < KCL; ++i) s += red[i];
        ws[OFF_LD + b] = s / (float)(KCL * (KCL - 1));
    }
    __syncthreads();
    red[lane] = (nrm[lane] > 0.0f) ? sqrtf(nrm[lane]) : 0.0f;
    __syncthreads();
    if (lane == 0) {
        float s = 0.0f;
        for (int i = 0; i < KCL; ++i) s += red[i];
        ws[OFF_LR + b] = s / (float)KCL;
    }
}

// ---------------------------------------------------------------------------
// Pass 2: L_var hinge to own centroid. Chunks walked in REVERSE global order
// so the L2-resident tail of pass 1 (~192MB of 256MB) is re-hit first.
// Non-temporal (last-use) loads for the streamed embedding.
// ---------------------------------------------------------------------------
__global__ void __launch_bounds__(TPB)
var_kernel(const float* __restrict__ emb, const int* __restrict__ labels,
           float* __restrict__ ws) {
    __shared__ float lmu[CHAN * KCL];      // [c][k]
    __shared__ float lv[KCL];

    const int g     = BATCH * NBLK - 1 - (int)blockIdx.x;   // reverse order
    const int b     = g / NBLK;
    const int chunk = g % NBLK;
    const int tid   = threadIdx.x;

    const float* gmu = ws + OFF_MU + (size_t)b * CHAN * KCL;
    for (int i = tid; i < CHAN * KCL; i += TPB) lmu[i] = gmu[i];
    if (tid < KCL) lv[tid] = 0.0f;
    __syncthreads();

    const int n0 = chunk * CHUNK + tid * PPT;
    const int* lp = labels + (size_t)b * NPTS + n0;
    v4i li0 = *(const v4i*)(lp);
    v4i li1 = *(const v4i*)(lp + 4);
    int lab[PPT] = {li0.x, li0.y, li0.z, li0.w, li1.x, li1.y, li1.z, li1.w};

    float d2[PPT];
#pragma unroll
    for (int p = 0; p < PPT; ++p) d2[p] = 0.0f;

    const float* base = emb + (size_t)b * CHAN * NPTS + n0;
    for (int c = 0; c < CHAN; ++c) {
        const float* xp = base + (size_t)c * NPTS;
        if (c + 1 < CHAN) __builtin_prefetch(xp + NPTS, 0, 1);
        v4f x0 = __builtin_nontemporal_load((const v4f*)(xp));
        v4f x1 = __builtin_nontemporal_load((const v4f*)(xp + 4));
        float xv[PPT] = {x0.x, x0.y, x0.z, x0.w, x1.x, x1.y, x1.z, x1.w};
        const float* murow = &lmu[c * KCL];
#pragma unroll
        for (int p = 0; p < PPT; ++p) {
            float diff = xv[p] - murow[lab[p]];
            d2[p] = fmaf(diff, diff, d2[p]);
        }
    }

#pragma unroll
    for (int p = 0; p < PPT; ++p) {
        float d = (d2[p] > 0.0f) ? sqrtf(d2[p]) : 0.0f;
        float t = d - DELTA_V;
        if (t > 0.0f) atomicAdd(&lv[lab[p]], t * t);
    }
    __syncthreads();
    if (tid < KCL) atomicAdd(&ws[OFF_V + b * KCL + tid], lv[tid]);
}

// ---------------------------------------------------------------------------
// Final scalar: mean_b( Lv_b + Ld_b + 0.001*Lr_b )
// ---------------------------------------------------------------------------
__global__ void __launch_bounds__(32)
finalize_kernel(const float* __restrict__ ws, float* __restrict__ out) {
    __shared__ float red[KCL];
    const int lane = threadIdx.x;
    float acc = 0.0f;
    for (int b = 0; b < BATCH; ++b) {
        float cnt = ws[OFF_CNT + b * KCL + lane];
        float vs  = ws[OFF_V   + b * KCL + lane];
        red[lane] = (cnt > 0.0f) ? (vs / cnt) : 0.0f;
        __syncthreads();
        if (lane == 0) {
            float s = 0.0f;
            for (int i = 0; i < KCL; ++i) s += red[i];
            acc += s / (float)KCL + ws[OFF_LD + b] + 0.001f * ws[OFF_LR + b];
        }
        __syncthreads();
    }
    if (lane == 0) out[0] = acc / (float)BATCH;
}

extern "C" void kernel_launch(void* const* d_in, const int* in_sizes, int n_in,
                              void* d_out, int out_size, void* d_ws, size_t ws_size,
                              hipStream_t stream) {
    const float* emb    = (const float*)d_in[0];   // (4,64,262144) f32
    const int*   labels = (const int*)d_in[1];     // (4,262144) i32
    float*       ws     = (float*)d_ws;
    float*       out    = (float*)d_out;

    init_ws_kernel<<<(WS_FLOATS + 255) / 256, 256, 0, stream>>>(ws);
    seg_sum_kernel<<<BATCH * NBLK, TPB, 0, stream>>>(emb, labels, ws);
    centroid_dist_kernel<<<BATCH, 32, 0, stream>>>(ws);
    var_kernel<<<BATCH * NBLK, TPB, 0, stream>>>(emb, labels, ws);
    finalize_kernel<<<1, 32, 0, stream>>>(ws, out);
}
